// DogeInnerFuncAttn_78778290144066
// MI455X (gfx1250) — matrix operand order repaired and checked by reading, not compiled, not run on hardware
//
#include <hip/hip_runtime.h>

// ---------------------------------------------------------------------------
// Types for CDNA5 WMMA (wave32, 16x16x32 bf16 -> f32)
// ---------------------------------------------------------------------------
typedef __attribute__((ext_vector_type(16))) __bf16 v16bf;
typedef __attribute__((ext_vector_type(8)))  float  v8f;

union Frag16 { v16bf v; uint4 u[2]; };

#define BM 128
#define BN 128
#define BKT 32
#define GEMM_THREADS 128   // 4 waves; each wave owns a 64x64 output tile

// ---------------------------------------------------------------------------
// CDNA5 async memory->LDS copy (ISA 15.18.3 op 98, tracked by ASYNCcnt).
// No VGPR staging: VDST carries the LDS byte offset (low 32 bits of the
// generic LDS pointer), VADDR the 64-bit global address. 16B per lane.
// ---------------------------------------------------------------------------
__device__ __forceinline__ void async_copy16(const __bf16* g, __bf16* l)
{
    unsigned lds_off = (unsigned)(unsigned long long)l;  // addr[31:0] = LDS offset
    unsigned long long ga = (unsigned long long)g;
    asm volatile("global_load_async_to_lds_b128 %0, %1, off"
                 :: "v"(lds_off), "v"(ga) : "memory");
}

__device__ __forceinline__ void wait_async0()
{
    asm volatile("s_wait_asynccnt 0x0" ::: "memory");
}

// ---------------------------------------------------------------------------
// Generic bf16 WMMA GEMM: C[M,N] = A[M,K] * Bt[N,K]^T   (f32 accumulate)
// A row-major (lda), Bt row-major along K (ldb), C row-major f32 (ldc).
// M,N multiples of 128; K multiple of 32. 128 threads = 4 waves in a 2x2
// grid; each wave computes 64x64 = 4x4 WMMA tiles per K-step:
//   16 v_wmma per 16 ds_load_b128 (fragments reused across 4 WMMAs each).
// Double-buffered LDS fed by async global->LDS DMA (ASYNCcnt), plus
// global_prefetch of the k+2 tile to warm GL2.
// ---------------------------------------------------------------------------
__global__ __launch_bounds__(GEMM_THREADS)
void gemm_bf16_wmma(const __bf16* __restrict__ A, int lda,
                    const __bf16* __restrict__ Bt, int ldb,
                    float* __restrict__ C, int ldc,
                    int M, int N, int Kd)
{
    __shared__ __bf16 As[2][BM * BKT];
    __shared__ __bf16 Bs[2][BN * BKT];

    const int tid  = threadIdx.x;
    const int bm   = blockIdx.y * BM;
    const int bn   = blockIdx.x * BN;
    const int wid  = tid >> 5;
    const int lane = tid & 31;
    const int half = lane >> 4;        // 0: lanes 0-15, 1: lanes 16-31
    const int l16  = lane & 15;
    const int wm   = (wid & 1) * 64;   // wave row offset in tile
    const int wn   = (wid >> 1) * 64;  // wave col offset in tile

    v8f acc[4][4] = {};

    const int nk = Kd / BKT;

    auto load_tiles = [&](int kt, int buf) {
        const int k0 = kt * BKT;
#pragma unroll
        for (int i = 0; i < 4; ++i) {
            int c   = tid + i * GEMM_THREADS;   // 0..511 -> 128 rows x 4 chunks
            int row = c >> 2;
            int col = (c & 3) * 8;              // 8 bf16 = 16B per chunk
            async_copy16(&A[(size_t)(bm + row) * lda + k0 + col],
                         &As[buf][row * BKT + col]);
            async_copy16(&Bt[(size_t)(bn + row) * ldb + k0 + col],
                         &Bs[buf][row * BKT + col]);
        }
    };

    load_tiles(0, 0);

    for (int kt = 0; kt < nk; ++kt) {
        wait_async0();          // this wave's async LDS writes have landed
        __syncthreads();        // all waves' writes visible
        if (kt + 1 < nk) load_tiles(kt + 1, (kt + 1) & 1);
        if (kt + 2 < nk) {
            // warm GL2 for the tile after next (global_prefetch_b8)
            const int k2 = (kt + 2) * BKT;
            __builtin_prefetch(&A[(size_t)(bm + tid) * lda + k2], 0, 1);
            __builtin_prefetch(&Bt[(size_t)(bn + tid) * ldb + k2], 0, 1);
        }
        const int buf = kt & 1;

        // A fragment (16x32 bf16, ISA 7.12.2): lane half selects K run of 8,
        // elements 0..7 = A[m][h*8 .. h*8+7], 8..15 = A[m][16+h*8 .. +7]
        Frag16 fa[4];
#pragma unroll
        for (int mt = 0; mt < 4; ++mt) {
            int base = (wm + mt * 16 + l16) * BKT + half * 8;
            fa[mt].u[0] = *(const uint4*)(&As[buf][base]);
            fa[mt].u[1] = *(const uint4*)(&As[buf][base + 16]);
        }
        // B fragment from [N][K] tile: elements 0..7 = Bt[n][h*8 .. h*8+7], 8..15 = Bt[n][16+h*8 .. +7]
        Frag16 fb[4];
#pragma unroll
        for (int nt = 0; nt < 4; ++nt) {
            int base = (wn + nt * 16 + l16) * BKT + half * 8;
            fb[nt].u[0] = *(const uint4*)(&Bs[buf][base]);
            fb[nt].u[1] = *(const uint4*)(&Bs[buf][base + 16]);
        }
#pragma unroll
        for (int mt = 0; mt < 4; ++mt)
#pragma unroll
            for (int nt = 0; nt < 4; ++nt)
                acc[mt][nt] = __builtin_amdgcn_wmma_f32_16x16x32_bf16(
                    false, fa[mt].v, false, fb[nt].v,
                    (short)0, acc[mt][nt], false, false);
    }

    // C/D layout (ISA 7.12.2): VGPR r -> M = r + 8*half, N = l16
#pragma unroll
    for (int mt = 0; mt < 4; ++mt) {
        int rowb = bm + wm + mt * 16 + half * 8;
#pragma unroll
        for (int nt = 0; nt < 4; ++nt) {
            int col = bn + wn + nt * 16 + l16;
#pragma unroll
            for (int r = 0; r < 8; ++r)
                C[(size_t)(rowb + r) * ldc + col] = acc[mt][nt][r];
        }
    }
}

// ---------------------------------------------------------------------------
// Elementwise f32 -> bf16
// ---------------------------------------------------------------------------
__global__ void cvt_bf16(const float* __restrict__ in, __bf16* __restrict__ out,
                         size_t n)
{
    size_t i = (size_t)blockIdx.x * 256 + threadIdx.x;
    size_t stride = (size_t)gridDim.x * 256;
    for (; i < n; i += stride) out[i] = (__bf16)in[i];
}

// Transpose + convert: out[c*rows + r] = bf16(in[r*cols + c])
__global__ void cvt_t_bf16(const float* __restrict__ in, __bf16* __restrict__ out,
                           int rows, int cols)
{
    int tx = threadIdx.x & 15;   // fast over r -> coalesced writes
    int ty = threadIdx.x >> 4;
    int r = blockIdx.x * 16 + tx;
    int c = blockIdx.y * 16 + ty;
    out[(size_t)c * rows + r] = (__bf16)in[(size_t)r * cols + c];
}

// ---------------------------------------------------------------------------
// RoPE on q (k is NOT roped in the reference), output bf16
// ---------------------------------------------------------------------------
__global__ __launch_bounds__(256)
void rope_q(const float* __restrict__ qf, const float* __restrict__ cosb,
            const float* __restrict__ sinb, __bf16* __restrict__ qb,
            int S_, int D_)
{
    int s = blockIdx.x;
    for (int c = threadIdx.x; c < D_; c += 256) {
        int d = c & 127;
        float q   = qf[(size_t)s * D_ + c];
        float rot = (d < 64) ? -qf[(size_t)s * D_ + c + 64]
                             :  qf[(size_t)s * D_ + c - 64];
        qb[(size_t)s * D_ + c] =
            (__bf16)(q * cosb[s * 128 + d] + rot * sinb[s * 128 + d]);
    }
}

// ---------------------------------------------------------------------------
// Top-4 per row (wave32 butterfly merge). One wave per row.
// ---------------------------------------------------------------------------
__device__ __forceinline__ void ins4(float v, int ix, float tv[4], int ti[4])
{
    if (v > tv[3]) {
        int pos = 3;
        while (pos > 0 && v > tv[pos - 1]) {
            tv[pos] = tv[pos - 1]; ti[pos] = ti[pos - 1]; --pos;
        }
        tv[pos] = v; ti[pos] = ix;
    }
}

__global__ __launch_bounds__(256)
void topk4_rows(const float* __restrict__ sim, int* __restrict__ idx, int ncols)
{
    int wid = threadIdx.x >> 5, lane = threadIdx.x & 31;
    int row = blockIdx.x * 8 + wid;
    const float* p = sim + (size_t)row * ncols;
    float tv[4] = {-3.4e38f, -3.4e38f, -3.4e38f, -3.4e38f};
    int   ti[4] = {-1, -1, -1, -1};
    for (int c = lane; c < ncols; c += 32) ins4(p[c], c, tv, ti);
#pragma unroll
    for (int off = 16; off >= 1; off >>= 1) {
        float ov[4]; int oi[4];
#pragma unroll
        for (int j = 0; j < 4; ++j) {
            ov[j] = __shfl_xor(tv[j], off, 32);
            oi[j] = __shfl_xor(ti[j], off, 32);
        }
#pragma unroll
        for (int j = 0; j < 4; ++j) ins4(ov[j], oi[j], tv, ti);
    }
    if (lane == 0) {
#pragma unroll
        for (int j = 0; j < 4; ++j) idx[row * 4 + j] = ti[j];
    }
}

// ---------------------------------------------------------------------------
// v[s,d] = hs[s,d] * sum_{hv,k} v_embed[idx[hv,s,k], d]; write transposed bf16
// v_t[d*S + s]  (so PV GEMM can consume it as Bt[N=d][K=s])
// ---------------------------------------------------------------------------
__global__ __launch_bounds__(256)
void gather_v(const float* __restrict__ hs, const float* __restrict__ v_embed,
              const int* __restrict__ idx, __bf16* __restrict__ v_t,
              int S_, int D_)
{
    __shared__ int ix[16];
    int s = blockIdx.x;
    if (threadIdx.x < 16) {
        int hv = threadIdx.x >> 2, j = threadIdx.x & 3;
        ix[threadIdx.x] = idx[((size_t)hv * S_ + s) * 4 + j];
    }
    __syncthreads();
    for (int d = threadIdx.x; d < D_; d += 256) {
        float sum = 0.f;
#pragma unroll
        for (int i = 0; i < 16; ++i)
            sum += v_embed[(size_t)ix[i] * D_ + d];
        v_t[(size_t)d * S_ + s] = (__bf16)(hs[(size_t)s * D_ + d] * sum);
    }
}

// ---------------------------------------------------------------------------
// Masked softmax row: P[s,t] = softmax_t( scores*1/sqrt(128) + mask ) -> bf16
// masked when t>s (causal) or attention_mask[t]*dynamic_mask[t]==0
// ---------------------------------------------------------------------------
__global__ __launch_bounds__(256)
void softmax_row(const float* __restrict__ sc, const float* __restrict__ amask,
                 const float* __restrict__ dmask, __bf16* __restrict__ P, int S_)
{
    __shared__ float red[256];
    int s = blockIdx.x;
    const float scale = 0.0883883476483184f; // 1/sqrt(128)
    const float NEG = -3.0e38f;
    const float* row = sc + (size_t)s * S_;

    float mx = NEG;
    for (int t = threadIdx.x; t < S_; t += 256) {
        float m = amask[t] * dmask[t];
        float x = (t <= s && m != 0.f) ? row[t] * scale : NEG;
        mx = fmaxf(mx, x);
    }
    red[threadIdx.x] = mx; __syncthreads();
    for (int o = 128; o > 0; o >>= 1) {
        if (threadIdx.x < o) red[threadIdx.x] = fmaxf(red[threadIdx.x], red[threadIdx.x + o]);
        __syncthreads();
    }
    mx = red[0]; __syncthreads();

    float sum = 0.f;
    for (int t = threadIdx.x; t < S_; t += 256) {
        float m = amask[t] * dmask[t];
        float x = (t <= s && m != 0.f) ? row[t] * scale : NEG;
        sum += __expf(x - mx);
    }
    red[threadIdx.x] = sum; __syncthreads();
    for (int o = 128; o > 0; o >>= 1) {
        if (threadIdx.x < o) red[threadIdx.x] += red[threadIdx.x + o];
        __syncthreads();
    }
    sum = red[0];
    float inv = 1.f / sum;
    for (int t = threadIdx.x; t < S_; t += 256) {
        float m = amask[t] * dmask[t];
        float x = (t <= s && m != 0.f) ? row[t] * scale : NEG;
        P[(size_t)s * S_ + t] = (__bf16)(__expf(x - mx) * inv);
    }
}

// ---------------------------------------------------------------------------
// Host-side orchestration
// ---------------------------------------------------------------------------
extern "C" void kernel_launch(void* const* d_in, const int* in_sizes, int n_in,
                              void* d_out, int out_size, void* d_ws, size_t ws_size,
                              hipStream_t stream)
{
    (void)in_sizes; (void)n_in; (void)out_size; (void)ws_size;
    const int S = 2048, D = 2048, H = 16, HD = 128, HV = 4, R = 128, NV = 2048, K = 4;

    const float* hs    = (const float*)d_in[0];
    const float* amask = (const float*)d_in[1];
    const float* cosb  = (const float*)d_in[2];
    const float* sinb  = (const float*)d_in[3];
    const float* dmask = (const float*)d_in[4];
    const float* Wq    = (const float*)d_in[5];
    const float* Wk    = (const float*)d_in[6];
    const float* Wvq   = (const float*)d_in[7];
    const float* vkeys = (const float*)d_in[8];
    const float* vemb  = (const float*)d_in[9];
    const float* Wo    = (const float*)d_in[10];
    float* out = (float*)d_out;

    char* w = (char*)d_ws;
    size_t off = 0;
    auto nxt = [&](size_t bytes) -> char* {
        char* p = w + off;
        off += (bytes + 255) & ~(size_t)255;
        return p;
    };
    __bf16* hs_bf   = (__bf16*)nxt((size_t)S * D * 2);
    __bf16* wq_t    = (__bf16*)nxt((size_t)D * D * 2);
    __bf16* wk_t    = (__bf16*)nxt((size_t)D * D * 2);
    __bf16* wvq_t   = (__bf16*)nxt((size_t)D * (HV * R) * 2);
    __bf16* wo_t    = (__bf16*)nxt((size_t)D * D * 2);
    __bf16* vkeys_t = (__bf16*)nxt((size_t)HV * NV * R * 2);
    float*  qf      = (float*) nxt((size_t)S * D * 4);
    float*  kf      = (float*) nxt((size_t)S * D * 4);
    __bf16* q_bf    = (__bf16*)nxt((size_t)S * D * 2);
    __bf16* k_bf    = (__bf16*)nxt((size_t)S * D * 2);
    float*  vq_f    = (float*) nxt((size_t)S * (HV * R) * 4);
    __bf16* vq_bf   = (__bf16*)nxt((size_t)S * (HV * R) * 2);
    float*  scoresf = (float*) nxt((size_t)S * S * 4);      // reused: sim & attn scores
    __bf16* P_bf    = (__bf16*)nxt((size_t)S * S * 2);
    __bf16* v_t     = (__bf16*)nxt((size_t)D * S * 2);
    int*    idxb    = (int*)   nxt((size_t)HV * S * K * 4);
    float*  ao_f    = (float*) nxt((size_t)S * D * 4);
    __bf16* ao_bf   = (__bf16*)nxt((size_t)S * D * 2);

    // 1) precision conversion / weight transposes
    cvt_bf16<<<4096, 256, 0, stream>>>(hs, hs_bf, (size_t)S * D);
    cvt_t_bf16<<<dim3(D / 16, D / 16), 256, 0, stream>>>(Wq, wq_t, D, D);
    cvt_t_bf16<<<dim3(D / 16, D / 16), 256, 0, stream>>>(Wk, wk_t, D, D);
    cvt_t_bf16<<<dim3(D / 16, (HV * R) / 16), 256, 0, stream>>>(Wvq, wvq_t, D, HV * R);
    cvt_t_bf16<<<dim3(D / 16, D / 16), 256, 0, stream>>>(Wo, wo_t, D, D);
    for (int hv = 0; hv < HV; ++hv)
        cvt_t_bf16<<<dim3(R / 16, NV / 16), 256, 0, stream>>>(
            vkeys + (size_t)hv * R * NV, vkeys_t + (size_t)hv * NV * R, R, NV);

    // 2) projections (WMMA GEMMs)
    gemm_bf16_wmma<<<dim3(D / BN, S / BM), GEMM_THREADS, 0, stream>>>(
        hs_bf, D, wq_t, D, qf, D, S, D, D);
    gemm_bf16_wmma<<<dim3(D / BN, S / BM), GEMM_THREADS, 0, stream>>>(
        hs_bf, D, wk_t, D, kf, D, S, D, D);
    gemm_bf16_wmma<<<dim3((HV * R) / BN, S / BM), GEMM_THREADS, 0, stream>>>(
        hs_bf, D, wvq_t, D, vq_f, HV * R, S, HV * R, D);

    rope_q<<<S, 256, 0, stream>>>(qf, cosb, sinb, q_bf, S, D);
    cvt_bf16<<<4096, 256, 0, stream>>>(kf, k_bf, (size_t)S * D);
    cvt_bf16<<<2048, 256, 0, stream>>>(vq_f, vq_bf, (size_t)S * (HV * R));

    // 3) router: sim GEMM + top-4 per HV head (reuse scores buffer)
    for (int hv = 0; hv < HV; ++hv) {
        gemm_bf16_wmma<<<dim3(NV / BN, S / BM), GEMM_THREADS, 0, stream>>>(
            vq_bf + hv * R, HV * R, vkeys_t + (size_t)hv * NV * R, R,
            scoresf, NV, S, NV, R);
        topk4_rows<<<S / 8, 256, 0, stream>>>(scoresf, idxb + (size_t)hv * S * K, NV);
    }
    gather_v<<<S, 256, 0, stream>>>(hs, vemb, idxb, v_t, S, D);

    // 4) attention, head by head (scores/P buffers reused; stream-ordered)
    for (int h = 0; h < H; ++h) {
        gemm_bf16_wmma<<<dim3(S / BN, S / BM), GEMM_THREADS, 0, stream>>>(
            q_bf + h * HD, D, k_bf + h * HD, D, scoresf, S, S, S, HD);
        softmax_row<<<S, 256, 0, stream>>>(scoresf, amask, dmask + (size_t)h * S, P_bf, S);
        gemm_bf16_wmma<<<dim3(HD / BN, S / BM), GEMM_THREADS, 0, stream>>>(
            P_bf, S, v_t + (size_t)h * HD * S, S, ao_f + h * HD, D, S, HD, S);
    }

    // 5) output projection
    cvt_bf16<<<4096, 256, 0, stream>>>(ao_f, ao_bf, (size_t)S * D);
    gemm_bf16_wmma<<<dim3(D / BN, S / BM), GEMM_THREADS, 0, stream>>>(
        ao_bf, D, wo_t, D, out, D, S, D, D);
}
